// PhyloDisentangler_27986006901263
// MI455X (gfx1250) — matrix operand order repaired and checked by reading, not compiled, not run on hardware
//
#include <hip/hip_runtime.h>
#include <hip/hip_bf16.h>
#include <math.h>

// ---------------------------------------------------------------------------
// PhyloDisentangler forward for MI455X (gfx1250, wave32, WMMA).
//
// Roofline: 1 GB fp32 weights @ 23.3 TB/s => ~44us floor; needs ~780 TFLOPS
// effective => bf16 WMMA (v_wmma_f32_16x16x32_bf16) with in-flight fp32->bf16
// conversion of the streamed weights, fp32 accumulation, LDS double-buffering
// and async global->LDS copies for the bf16 activations (ASYNCcnt path).
// ---------------------------------------------------------------------------

typedef __attribute__((ext_vector_type(16))) __bf16 v16bf;
typedef __attribute__((ext_vector_type(8)))  float  v8f;

#define B_SZ     64
#define RESQ     256      // RES*RES
#define FLAT_IN  16384    // 64*256
#define FLAT_Z   4096     // 64*16*4
#define N_CODES  1024
#define EMB      64

// ---- CDNA5 async global->LDS path (ASYNCcnt) ------------------------------
// Probe-verified: param0 = AS(1) 'int __vector(4)*' (global src),
// param1 = AS(3) vector ptr (LDS dst), then imm offset, imm cpol.
#if defined(__has_builtin)
#if __has_builtin(__builtin_amdgcn_global_load_async_to_lds_b128) && \
    __has_builtin(__builtin_amdgcn_s_wait_asynccnt)
#define USE_ASYNC_LDS 1
#endif
#endif

#ifdef USE_ASYNC_LDS
typedef int async_b128_t __attribute__((__vector_size__(4 * sizeof(int))));
typedef __attribute__((address_space(1))) async_b128_t* g_b128_ptr;
typedef __attribute__((address_space(3))) async_b128_t* l_b128_ptr;
#define ASYNC_COPY_B128(dst_lds, src_glb) \
  __builtin_amdgcn_global_load_async_to_lds_b128( \
      (g_b128_ptr)(src_glb), (l_b128_ptr)(dst_lds), 0, 0)
#define ASYNC_WAIT0() __builtin_amdgcn_s_wait_asynccnt(0)
#endif

// ---- WMMA fragment helpers (layouts per CDNA5 ISA 7.12.2) ------

// A fragment: 16x32 bf16 tile in LDS, row stride 40 elements.
// lane = half*16 + r holds row M=r; VGPR v holds K = {half*8+2v, +1} (v<4)
// or K = {16+half*8+2(v-4), +1} (v>=4).
__device__ __forceinline__ v16bf load_afrag(const __bf16* base, int half, int r) {
  v16bf a;
  const __bf16* rp = base + r * 40;
#pragma unroll
  for (int v = 0; v < 8; ++v) {
    int kb = (v < 4) ? (half * 8 + 2 * v) : (16 + half * 8 + 2 * (v - 4));
    reinterpret_cast<unsigned*>(&a)[v] =
        *reinterpret_cast<const unsigned*>(rp + kb);
  }
  return a;
}

// B fragment: 32x16 bf16 tile in LDS, row stride 136 elements (K-major rows).
// lane = half*16 + n holds column N=n; element s holds K = half*8+s (s<8)
// or K = 16+half*8+(s-8) (s>=8).
__device__ __forceinline__ v16bf load_bfrag(const __bf16* base, int lane, int nc) {
  v16bf b;
  const __bf16* cp = base + nc + (lane & 15);
  const int h = lane >> 4;
#pragma unroll
  for (int s = 0; s < 16; ++s) {
    int k = (s < 8) ? (h * 8 + s) : (16 + h * 8 + (s - 8));
    b[s] = cp[k * 136];
  }
  return b;
}

__device__ __forceinline__ float silu_f(float x) {
  return x / (1.f + __expf(-x));
}

// ---------------------------------------------------------------------------
// GEMM: Z(64 x N) += A(64 x K, bf16) * W(K x N, fp32->bf16).  Split-K via
// fp32 global atomics; Z must be pre-filled with the bias.  Double-buffered
// LDS: next tile's global traffic is issued before the current tile's WMMAs.
// All fragments are preloaded so the 4 WMMAs issue back-to-back (one DS
// drain per iteration instead of one per WMMA).
// grid = (N/128, K/kPerSplit), block = 256 (8 waves; wave w owns N strip w*16).
// ---------------------------------------------------------------------------
__global__ __launch_bounds__(256)
void gemm_a64_kernel(const __bf16* __restrict__ A, const float* __restrict__ W,
                     float* __restrict__ Z, int K, int N, int kPerSplit) {
  __shared__ __bf16 As[2][64][40];
  __shared__ __bf16 Bs[2][32][136];
  const int t = threadIdx.x;
  const int wave = t >> 5, lane = t & 31, half = lane >> 4, r = lane & 15;
  const int n0 = blockIdx.x * 128;
  const int kStart = blockIdx.y * kPerSplit;
  const int nIter = kPerSplit >> 5;

  // Per-thread streaming pointers (no per-iteration 64-bit muls).
  const int arow = t >> 2, aseg = (t & 3) * 8;      // A: 64 rows x 4x8 bf16
  const __bf16* ap = A + (long)arow * K + kStart + aseg;
  const float* wp = W + (long)(kStart + wave) * N + n0 + lane * 4;  // +i*8 rows
  const long wRow8 = (long)8 * N;

  v8f acc[4];
#pragma unroll
  for (int i = 0; i < 4; ++i) { v8f z = {}; acc[i] = z; }

  float4 wreg[4];
  uint4 areg;

  // ---- prologue: tile 0 ----
#pragma unroll
  for (int i = 0; i < 4; ++i) wreg[i] = *reinterpret_cast<const float4*>(wp + i * wRow8);
#ifdef USE_ASYNC_LDS
  ASYNC_COPY_B128(&As[0][arow][aseg], ap);
#else
  areg = *reinterpret_cast<const uint4*>(ap);
#endif
#pragma unroll
  for (int i = 0; i < 4; ++i) {
    __bf16* bd = &Bs[0][wave + i * 8][lane * 4];
    bd[0] = (__bf16)wreg[i].x; bd[1] = (__bf16)wreg[i].y;
    bd[2] = (__bf16)wreg[i].z; bd[3] = (__bf16)wreg[i].w;
  }
#ifndef USE_ASYNC_LDS
  *reinterpret_cast<uint4*>(&As[0][arow][aseg]) = areg;
#endif
  ap += 32;
  wp += 4 * wRow8;
#ifdef USE_ASYNC_LDS
  ASYNC_WAIT0();
#endif
  __syncthreads();

  for (int it = 0; it < nIter; ++it) {
    const int cur = it & 1, nxt = cur ^ 1;
    const bool more = (it + 1) < nIter;

    // Issue next tile's global traffic before this tile's compute.
    if (more) {
#pragma unroll
      for (int i = 0; i < 4; ++i) wreg[i] = *reinterpret_cast<const float4*>(wp + i * wRow8);
#ifdef USE_ASYNC_LDS
      ASYNC_COPY_B128(&As[nxt][arow][aseg], ap);
#else
      areg = *reinterpret_cast<const uint4*>(ap);
#endif
      ap += 32;
      wp += 4 * wRow8;
      if ((it + 2) < nIter)  // hint the tile after next (global_prefetch_b8)
        __builtin_prefetch(wp + wave * wRow8 / 8, 0, 3);
    }

    // Preload all fragments, then issue the 4 WMMAs back-to-back.
    v16bf bf = load_bfrag(&Bs[cur][0][0], lane, wave * 16);
    v16bf af[4];
#pragma unroll
    for (int mt = 0; mt < 4; ++mt)
      af[mt] = load_afrag(&As[cur][mt * 16][0], half, r);
#pragma unroll
    for (int mt = 0; mt < 4; ++mt)
      acc[mt] = __builtin_amdgcn_wmma_f32_16x16x32_bf16(
          false, af[mt], false, bf, (short)0, acc[mt], false, false);

    // Land next tile in the other LDS buffer.
    if (more) {
#pragma unroll
      for (int i = 0; i < 4; ++i) {
        __bf16* bd = &Bs[nxt][wave + i * 8][lane * 4];
        bd[0] = (__bf16)wreg[i].x; bd[1] = (__bf16)wreg[i].y;
        bd[2] = (__bf16)wreg[i].z; bd[3] = (__bf16)wreg[i].w;
      }
#ifndef USE_ASYNC_LDS
      *reinterpret_cast<uint4*>(&As[nxt][arow][aseg]) = areg;
#endif
    }
#ifdef USE_ASYNC_LDS
    ASYNC_WAIT0();
#endif
    __syncthreads();
  }

#pragma unroll
  for (int mt = 0; mt < 4; ++mt)
#pragma unroll
    for (int v = 0; v < 8; ++v)
      atomicAdd(&Z[(long)(mt * 16 + v + 8 * half) * N + n0 + wave * 16 + r],
                acc[mt][v]);
}

// ---------------------------------------------------------------------------
// 1x1 conv as GEMM over (b,hw) rows: Out[b][n][hw] = act( sum_k A[b][k][hw]
// * W[k][n] + bias[n] ).  A source is channel-major per sample (x or the
// hout_p|hout_n concat, selected by kSplit), optional SiLU on input/output.
// Epilogue stages each wave's 16x16 tile through LDS for coalesced b128
// stores.  grid = (256, N/128), block = 256.
// ---------------------------------------------------------------------------
__global__ __launch_bounds__(256)
void conv_gemm_kernel(const float* __restrict__ A0, const float* __restrict__ A1,
                      int kSplit, int sampleStride,
                      const float* __restrict__ W, const float* __restrict__ bias,
                      float* __restrict__ Out, int K, int N,
                      int siluIn, int siluOut) {
  __shared__ __bf16 As[64][40];
  __shared__ __bf16 Bs[32][136];
  __shared__ float Ostage[8][16][20];
  const int t = threadIdx.x;
  const int wave = t >> 5, lane = t & 31, half = lane >> 4, r = lane & 15;
  const int b = blockIdx.x >> 2;
  const int hw0 = (blockIdx.x & 3) * 64;
  const int n0 = blockIdx.y * 128;
  const int kk = t >> 3;
  const int hs = (t & 7) * 8;

  v8f acc[4];
#pragma unroll
  for (int i = 0; i < 4; ++i) { v8f z = {}; acc[i] = z; }

  for (int kb = 0; kb < K; kb += 32) {
    // ---- batched load phase (all global loads in flight together) ----
    const int k = kb + kk;
    const float* src = (k < kSplit)
        ? (A0 + (long)b * sampleStride + (long)k * RESQ)
        : (A1 + (long)b * sampleStride + (long)(k - kSplit) * RESQ);
    const float4* pa = reinterpret_cast<const float4*>(src + hw0 + hs);
    float4 a0 = pa[0], a1 = pa[1];
    float4 w[4];
#pragma unroll
    for (int i = 0; i < 4; ++i)
      w[i] = *reinterpret_cast<const float4*>(
          W + (long)(kb + wave + i * 8) * N + n0 + lane * 4);

    // ---- store phase (fp32 -> bf16, optional SiLU on A) ----
    float av[8] = {a0.x, a0.y, a0.z, a0.w, a1.x, a1.y, a1.z, a1.w};
#pragma unroll
    for (int j = 0; j < 8; ++j) {
      float v = av[j];
      if (siluIn) v = silu_f(v);
      As[hs + j][kk] = (__bf16)v;  // transposed: LDS is [hw][k]
    }
#pragma unroll
    for (int i = 0; i < 4; ++i) {
      __bf16* bd = &Bs[wave + i * 8][lane * 4];
      bd[0] = (__bf16)w[i].x; bd[1] = (__bf16)w[i].y;
      bd[2] = (__bf16)w[i].z; bd[3] = (__bf16)w[i].w;
    }
    __syncthreads();

    // Preload all fragments, then issue the 4 WMMAs back-to-back.
    v16bf bf = load_bfrag(&Bs[0][0], lane, wave * 16);
    v16bf af[4];
#pragma unroll
    for (int mt = 0; mt < 4; ++mt)
      af[mt] = load_afrag(&As[mt * 16][0], half, r);
#pragma unroll
    for (int mt = 0; mt < 4; ++mt)
      acc[mt] = __builtin_amdgcn_wmma_f32_16x16x32_bf16(
          false, af[mt], false, bf, (short)0, acc[mt], false, false);
    __syncthreads();
  }

  // ---- epilogue: bias (+SiLU), stage 16x16 per wave, coalesced stores ----
  const int n = n0 + wave * 16 + r;
  const float bn = bias[n];
  const int nr = lane >> 1, cseg = (lane & 1) * 8;
#pragma unroll
  for (int mt = 0; mt < 4; ++mt) {
#pragma unroll
    for (int v = 0; v < 8; ++v) {
      float val = acc[mt][v] + bn;
      if (siluOut) val = silu_f(val);
      Ostage[wave][r][v + 8 * half] = val;  // [n_local][m_local]
    }
    // wave-private LDS transpose: lane writes 8 contiguous hw of one n row.
    float4 o0 = *reinterpret_cast<const float4*>(&Ostage[wave][nr][cseg]);
    float4 o1 = *reinterpret_cast<const float4*>(&Ostage[wave][nr][cseg + 4]);
    long obase = ((long)b * N + n0 + wave * 16 + nr) * RESQ + hw0 + mt * 16 + cseg;
    *reinterpret_cast<float4*>(Out + obase) = o0;
    *reinterpret_cast<float4*>(Out + obase + 4) = o1;
  }
}

// ---------------------------------------------------------------------------
// Per-sample LayerNorm stats over 16384 contiguous floats (64ch x 256 hw).
// grid = 128 (64 samples x {phylo, img}).  stats[s] = {mean, rsqrt(var+eps)}.
// ---------------------------------------------------------------------------
__global__ __launch_bounds__(256)
void ln_stats_kernel(const float* __restrict__ h, float* __restrict__ stats) {
  int s = blockIdx.x, b = s >> 1, hf = s & 1;
  const float* base = h + ((long)b * 128 + hf * 64) * RESQ;
  float sum = 0.f, sq = 0.f;
  for (int i = threadIdx.x; i < FLAT_IN; i += 256) {
    float v = base[i];
    sum += v;
    sq += v * v;
  }
  __shared__ float s1[256], s2[256];
  s1[threadIdx.x] = sum;
  s2[threadIdx.x] = sq;
  __syncthreads();
  for (int o = 128; o > 0; o >>= 1) {
    if (threadIdx.x < o) {
      s1[threadIdx.x] += s1[threadIdx.x + o];
      s2[threadIdx.x] += s2[threadIdx.x + o];
    }
    __syncthreads();
  }
  if (threadIdx.x == 0) {
    float m = s1[0] / (float)FLAT_IN;
    float var = s2[0] / (float)FLAT_IN - m * m;
    stats[s * 2] = m;
    stats[s * 2 + 1] = rsqrtf(var + 1e-5f);
  }
}

// Normalize + affine + bf16 cast into GEMM A layout a[b][16384].
__global__ __launch_bounds__(256)
void normalize_kernel(const float* __restrict__ h, const float* __restrict__ stats,
                      const float* __restrict__ g, const float* __restrict__ bt,
                      __bf16* __restrict__ a, int hf) {
  int idx0 = (blockIdx.x * 256 + threadIdx.x) * 4;
#pragma unroll
  for (int j = 0; j < 4; ++j) {
    int idx = idx0 + j;
    int b = idx >> 14, col = idx & 16383;
    float m = stats[(b * 2 + hf) * 2];
    float rv = stats[(b * 2 + hf) * 2 + 1];
    float v = h[((long)b * 128 + hf * 64) * RESQ + col];
    a[(long)b * FLAT_IN + col] = (__bf16)((v - m) * rv * g[col] + bt[col]);
  }
}

// Fill an accumulator buffer with a broadcast bias row.
__global__ void fill_bias_kernel(float* __restrict__ dst,
                                 const float* __restrict__ bias,
                                 int cols, int total) {
  int i = blockIdx.x * 256 + threadIdx.x;
  if (i < total) dst[i] = bias[i & (cols - 1)];
}

// Precompute |codebook_j|^2 and zero the loss accumulator.
__global__ void prep_kernel(const float* __restrict__ cb, float* __restrict__ cb2,
                            float* __restrict__ loss) {
  int j = blockIdx.x * 256 + threadIdx.x;
  if (j < N_CODES) {
    float s = 0.f;
#pragma unroll 8
    for (int e = 0; e < EMB; ++e) {
      float v = cb[j * EMB + e];
      s += v * v;
    }
    cb2[j] = s;
  }
  if (j == 0) *loss = 0.f;
}

// ---------------------------------------------------------------------------
// VQ: rows are (b, c, l) -> 64-dim vectors z[b][e*64 + c*4 + l] over e.
// One wave per row; lane covers 32 codes; argmin d = |c|^2 - 2 z.c via
// __shfl_xor over wave32.  Writes bf16 zq and accumulates sum (zq - z)^2.
// ---------------------------------------------------------------------------
__global__ __launch_bounds__(256)
void quantize_kernel(const float* __restrict__ z, const float* __restrict__ cb,
                     const float* __restrict__ cb2, __bf16* __restrict__ zq,
                     float* __restrict__ lossAcc) {
  __shared__ float zrow[8][64];
  const int wave = threadIdx.x >> 5, lane = threadIdx.x & 31;
  const int row = blockIdx.x * 8 + wave;  // 0..4095
  const int b = row >> 6, off = row & 63;
  const float* zb = z + (long)b * FLAT_Z + off;
  zrow[wave][lane] = zb[lane * 64];
  zrow[wave][lane + 32] = zb[(lane + 32) * 64];
  __syncthreads();

  float zr[64];
#pragma unroll
  for (int e = 0; e < EMB; ++e) zr[e] = zrow[wave][e];

  float best = 3.4e38f;
  int bidx = 0;
  for (int j = lane; j < N_CODES; j += 32) {
    const float* c = cb + j * EMB;
    float dot = 0.f;
#pragma unroll
    for (int e = 0; e < EMB; ++e) dot += zr[e] * c[e];
    float d = cb2[j] - 2.f * dot;
    if (d < best) { best = d; bidx = j; }
  }
#pragma unroll
  for (int m = 16; m > 0; m >>= 1) {
    float ob = __shfl_xor(best, m, 32);
    int oi = __shfl_xor(bidx, m, 32);
    if (ob < best || (ob == best && oi < bidx)) { best = ob; bidx = oi; }
  }

  const float* cv = cb + bidx * EMB;
  float lsum = 0.f;
  {
    float c0 = cv[lane];
    float d0 = c0 - zr[lane];
    lsum += d0 * d0;
    zq[(long)b * FLAT_Z + lane * 64 + off] = (__bf16)c0;
    float c1 = cv[lane + 32];
    float d1 = c1 - zr[lane + 32];
    lsum += d1 * d1;
    zq[(long)b * FLAT_Z + (lane + 32) * 64 + off] = (__bf16)c1;
  }
#pragma unroll
  for (int m = 16; m > 0; m >>= 1) lsum += __shfl_xor(lsum, m, 32);
  if (lane == 0) atomicAdd(lossAcc, lsum);
}

// quant_loss = (1+beta) * (sum_p + sum_n) / (B*EMBED*CODES*LEVELS)
__global__ void finalize_kernel(const float* __restrict__ loss,
                                float* __restrict__ out) {
  if (threadIdx.x == 0) out[0] = 1.25f * (*loss) / 262144.f;
}

// ---------------------------------------------------------------------------
extern "C" void kernel_launch(void* const* d_in, const int* in_sizes, int n_in,
                              void* d_out, int out_size, void* d_ws, size_t ws_size,
                              hipStream_t stream) {
  const float* x          = (const float*)d_in[0];
  const float* conv_in_w  = (const float*)d_in[1];
  const float* conv_in_b  = (const float*)d_in[2];
  const float* ln_p_g     = (const float*)d_in[3];
  const float* ln_p_b     = (const float*)d_in[4];
  const float* W_in_p     = (const float*)d_in[5];
  const float* b_in_p     = (const float*)d_in[6];
  const float* ln_n_g     = (const float*)d_in[7];
  const float* ln_n_b     = (const float*)d_in[8];
  const float* W_in_n     = (const float*)d_in[9];
  const float* b_in_n     = (const float*)d_in[10];
  const float* codebook   = (const float*)d_in[11];
  const float* W_out_p    = (const float*)d_in[12];
  const float* b_out_p    = (const float*)d_in[13];
  const float* W_out_n    = (const float*)d_in[14];
  const float* b_out_n    = (const float*)d_in[15];
  const float* conv_out_w = (const float*)d_in[16];
  const float* conv_out_b = (const float*)d_in[17];
  float* out = (float*)d_out;

  char* ws = (char*)d_ws;
  float*  h      = (float*)(ws + 0);            // 64*128*256 f32   (8 MB)
  __bf16* a_p    = (__bf16*)(ws + 8388608);     // 64*16384 bf16    (2 MB)
  __bf16* a_n    = (__bf16*)(ws + 10485760);    // 64*16384 bf16    (2 MB)
  float*  z_p    = (float*)(ws + 12582912);     // 64*4096 f32      (1 MB)
  float*  z_n    = (float*)(ws + 13631488);     // 64*4096 f32      (1 MB)
  __bf16* zq_p   = (__bf16*)(ws + 14680064);    // 64*4096 bf16     (0.5 MB)
  __bf16* zq_n   = (__bf16*)(ws + 15204352);    // 64*4096 bf16     (0.5 MB)
  float*  hout_p = (float*)(ws + 15728640);     // 64*16384 f32     (4 MB)
  float*  hout_n = (float*)(ws + 19922944);     // 64*16384 f32     (4 MB)
  float*  stats  = (float*)(ws + 24117248);     // 128*2 f32
  float*  cb2    = (float*)(ws + 24118272);     // 1024 f32
  float*  lossA  = (float*)(ws + 24122368);     // 1 f32

  // 0) prep: |cb|^2, zero loss; pre-fill GEMM accumulators with biases.
  prep_kernel<<<4, 256, 0, stream>>>(codebook, cb2, lossA);
  fill_bias_kernel<<<1024, 256, 0, stream>>>(z_p, b_in_p, FLAT_Z, B_SZ * FLAT_Z);
  fill_bias_kernel<<<1024, 256, 0, stream>>>(z_n, b_in_n, FLAT_Z, B_SZ * FLAT_Z);
  fill_bias_kernel<<<4096, 256, 0, stream>>>(hout_p, b_out_p, FLAT_IN, B_SZ * FLAT_IN);
  fill_bias_kernel<<<4096, 256, 0, stream>>>(hout_n, b_out_n, FLAT_IN, B_SZ * FLAT_IN);

  // 1) conv_in + SiLU: (16384 x 256)x(256 x 128) -> h[b][128][256]
  conv_gemm_kernel<<<dim3(256, 1), 256, 0, stream>>>(
      x, x, 256, 256 * RESQ, conv_in_w, conv_in_b, h, 256, 128, 0, 1);

  // 2) LayerNorm stats + normalize into bf16 GEMM A operands.
  ln_stats_kernel<<<128, 256, 0, stream>>>(h, stats);
  normalize_kernel<<<1024, 256, 0, stream>>>(h, stats, ln_p_g, ln_p_b, a_p, 0);
  normalize_kernel<<<1024, 256, 0, stream>>>(h, stats, ln_n_g, ln_n_b, a_n, 1);

  // 3) z = LN(h) @ W_in + b  (M=64, K=16384, N=4096), split-K=8.
  gemm_a64_kernel<<<dim3(32, 8), 256, 0, stream>>>(a_p, W_in_p, z_p, FLAT_IN, FLAT_Z, 2048);
  gemm_a64_kernel<<<dim3(32, 8), 256, 0, stream>>>(a_n, W_in_n, z_n, FLAT_IN, FLAT_Z, 2048);

  // 4) vector quantization + loss.
  quantize_kernel<<<512, 256, 0, stream>>>(z_p, codebook, cb2, zq_p, lossA);
  quantize_kernel<<<512, 256, 0, stream>>>(z_n, codebook, cb2, zq_n, lossA);

  // 5) hout = zq @ W_out + b  (M=64, K=4096, N=16384), split-K=4.
  gemm_a64_kernel<<<dim3(128, 4), 256, 0, stream>>>(zq_p, W_out_p, hout_p, FLAT_Z, FLAT_IN, 1024);
  gemm_a64_kernel<<<dim3(128, 4), 256, 0, stream>>>(zq_n, W_out_n, hout_n, FLAT_Z, FLAT_IN, 1024);

  // 6) conv_out over SiLU(concat(hout_p, hout_n)): (16384x128)x(128x256) -> out.
  conv_gemm_kernel<<<dim3(256, 2), 256, 0, stream>>>(
      hout_p, hout_n, 64, FLAT_IN, conv_out_w, conv_out_b, out, 128, 256, 1, 0);

  // 7) quant_loss scalar appended after the 64*256*16*16 output tensor.
  finalize_kernel<<<1, 32, 0, stream>>>(lossA, out + 4194304);
}
